// TransformerBlock_13383118094837
// MI455X (gfx1250) — matrix operand order repaired and checked by reading, not compiled, not run on hardware
//
#include <hip/hip_runtime.h>
#include <hip/hip_bf16.h>

#define Bn  4
#define Sn  1024
#define Hn  1024
#define NHn 16
#define Dn  64
#define En  8
#define Fn  4096
#define Tn  (Bn * Sn)
#define EPSF 1e-5f

typedef __attribute__((ext_vector_type(16))) __bf16 v16bf;
typedef __attribute__((ext_vector_type(8)))  __bf16 v8bf;
typedef __attribute__((ext_vector_type(8)))  float  v8f;

__device__ __forceinline__ unsigned short f2bf(float f) {
    union { float f; unsigned int u; } x; x.f = f;
    unsigned int r = (x.u + 0x7FFFu + ((x.u >> 16) & 1u)) >> 16;
    return (unsigned short)r;
}

// A-fragment (16x32 bf16): lane m (0-15) holds row m, K = {0..7,16..23}; lane m+16 K = {8..15,24..31}
__device__ __forceinline__ v16bf frag_a(const unsigned short* base, int ld, int lane) {
    int r = lane & 15, hi = (lane >> 4) & 1;
    const unsigned short* p = base + (long)r * ld + hi * 8;
    v8bf lo = *(const v8bf*)p;
    v8bf hv = *(const v8bf*)(p + 16);
    return __builtin_shufflevector(lo, hv, 0,1,2,3,4,5,6,7,8,9,10,11,12,13,14,15);
}

// B-fragment (32x16 bf16), B pre-transposed so "rows"=N columns: lane n holds col n, K = {0..7,16..23} (hi=0) / {8..15,24..31} (hi=1)
__device__ __forceinline__ v16bf frag_b(const unsigned short* base, int ld, int lane) {
    int r = lane & 15, hi = (lane >> 4) & 1;
    const unsigned short* p = base + (long)r * ld + hi * 8;
    v8bf lo = *(const v8bf*)p;
    v8bf hv = *(const v8bf*)(p + 16);
    return __builtin_shufflevector(lo, hv, 0,1,2,3,4,5,6,7,8,9,10,11,12,13,14,15);
}

__device__ __forceinline__ v8f wmma_bf16(v16bf a, v16bf b, v8f c) {
    return __builtin_amdgcn_wmma_f32_16x16x32_bf16(false, a, false, b, (short)0, c, false, false);
}

// -------- f32 [K,N] -> bf16 [N,K] transpose+convert (batched over blockIdx.z) --------
__global__ void k_transpose_cvt(const float* __restrict__ in, unsigned short* __restrict__ out,
                                int K, int N) {
    __shared__ float tile[32][33];
    long zoff = (long)blockIdx.z * K * N;
    in += zoff; out += zoff;
    int bk = blockIdx.x * 32, bn = blockIdx.y * 32;
    for (int i = threadIdx.y; i < 32; i += 8)
        tile[i][threadIdx.x] = in[(long)(bk + i) * N + bn + threadIdx.x];
    __syncthreads();
    for (int i = threadIdx.y; i < 32; i += 8)
        out[(long)(bn + i) * K + bk + threadIdx.x] = f2bf(tile[threadIdx.x][i]);
}

// -------- LayerNorm f32 row -> bf16 row --------
__global__ void k_ln_bf16(const float* __restrict__ x, const float* __restrict__ g,
                          const float* __restrict__ bt, unsigned short* __restrict__ o) {
    int t = blockIdx.x, tid = threadIdx.x;
    const float* row = x + (long)t * Hn;
    float4 v = ((const float4*)row)[tid];
    __shared__ float r1[256], r2[256];
    r1[tid] = v.x + v.y + v.z + v.w;
    r2[tid] = v.x * v.x + v.y * v.y + v.z * v.z + v.w * v.w;
    __syncthreads();
    for (int st = 128; st > 0; st >>= 1) {
        if (tid < st) { r1[tid] += r1[tid + st]; r2[tid] += r2[tid + st]; }
        __syncthreads();
    }
    float mu = r1[0] / Hn;
    float inv = rsqrtf(r2[0] / Hn - mu * mu + EPSF);
    int h0 = tid * 4;
    float n0 = (v.x - mu) * inv * g[h0 + 0] + bt[h0 + 0];
    float n1 = (v.y - mu) * inv * g[h0 + 1] + bt[h0 + 1];
    float n2 = (v.z - mu) * inv * g[h0 + 2] + bt[h0 + 2];
    float n3 = (v.w - mu) * inv * g[h0 + 3] + bt[h0 + 3];
    ushort4 p; p.x = f2bf(n0); p.y = f2bf(n1); p.z = f2bf(n2); p.w = f2bf(n3);
    *(ushort4*)&o[(long)t * Hn + h0] = p;
}

// -------- generic WMMA GEMM: C[M,N] = A[M,K](bf16) @ Bt[N,K](bf16)^T + bias --------
// MODE 0: bf16 out, scaled.  MODE 1: bf16 out transposed as [B,NH,D,S] (for V).  MODE 2: f32 out + resid.
template <int MODE>
__global__ __launch_bounds__(32) void k_gemm(const unsigned short* __restrict__ A,
                                             const unsigned short* __restrict__ Bt,
                                             const float* __restrict__ bias,
                                             const float* __restrict__ resid, float scale,
                                             unsigned short* __restrict__ outb,
                                             float* __restrict__ outf, int M, int N, int K) {
    int lane = threadIdx.x;
    int m0 = blockIdx.x * 32, n0 = blockIdx.y * 32;
    v8f acc[2][2] = {};
    const unsigned short* a0p = A + (long)m0 * K;
    const unsigned short* a1p = A + (long)(m0 + 16) * K;
    const unsigned short* b0p = Bt + (long)n0 * K;
    const unsigned short* b1p = Bt + (long)(n0 + 16) * K;
    for (int k = 0; k < K; k += 32) {
        v16bf a0 = frag_a(a0p + k, K, lane);
        v16bf a1 = frag_a(a1p + k, K, lane);
        v16bf b0 = frag_b(b0p + k, K, lane);
        v16bf b1 = frag_b(b1p + k, K, lane);
        acc[0][0] = wmma_bf16(a0, b0, acc[0][0]);
        acc[0][1] = wmma_bf16(a0, b1, acc[0][1]);
        acc[1][0] = wmma_bf16(a1, b0, acc[1][0]);
        acc[1][1] = wmma_bf16(a1, b1, acc[1][1]);
    }
    int r15 = lane & 15, hi = (lane >> 4) & 1;
    for (int i = 0; i < 2; i++)
        for (int j = 0; j < 2; j++) {
            int col = n0 + j * 16 + r15;
            float bv = bias ? bias[col] : 0.f;
            for (int r = 0; r < 8; r++) {
                int row = m0 + i * 16 + r + hi * 8;
                float val = acc[i][j][r] + bv;
                if (MODE == 0) {
                    outb[(long)row * N + col] = f2bf(val * scale);
                } else if (MODE == 1) {
                    int b_ = row >> 10, s_ = row & (Sn - 1);
                    int nh = col >> 6, d = col & 63;
                    outb[((long)(b_ * NHn + nh) * Dn + d) * Sn + s_] = f2bf(val);
                } else {
                    outf[(long)row * N + col] = val + resid[(long)row * N + col];
                }
            }
        }
}

// -------- flash attention: one 16-query tile per wave --------
__global__ __launch_bounds__(32) void k_attn(const unsigned short* __restrict__ q,
                                             const unsigned short* __restrict__ kb,
                                             const unsigned short* __restrict__ vt,
                                             unsigned short* __restrict__ ob) {
    int lane = threadIdx.x;
    int qt = blockIdx.x, nh = blockIdx.y, b = blockIdx.z;
    int t0 = b * Sn + qt * 16;
    int r15 = lane & 15, hi = (lane >> 4) & 1;
    const unsigned short* qbase = q + (long)t0 * Hn + nh * Dn;
    v16bf aq0 = frag_a(qbase, Hn, lane);
    v16bf aq1 = frag_a(qbase + 32, Hn, lane);
    v8f acc[4] = {};
    float mrow[8], lrow[8];
    for (int r = 0; r < 8; r++) { mrow[r] = -3.0e38f; lrow[r] = 0.f; }
    __shared__ __align__(16) unsigned short lds_p[16 * 32];
    const unsigned short* vbase = vt + (long)(b * NHn + nh) * Dn * Sn;
    for (int kc = 0; kc < Sn; kc += 32) {
        const unsigned short* kbase = kb + (long)(b * Sn + kc) * Hn + nh * Dn;
        v8f s0 = {}, s1 = {};
        {
            v16bf b00 = frag_b(kbase, Hn, lane);
            v16bf b01 = frag_b(kbase + 32, Hn, lane);
            s0 = wmma_bf16(aq0, b00, s0);
            s0 = wmma_bf16(aq1, b01, s0);
            v16bf b10 = frag_b(kbase + 16 * Hn, Hn, lane);
            v16bf b11 = frag_b(kbase + 16 * Hn + 32, Hn, lane);
            s1 = wmma_bf16(aq0, b10, s1);
            s1 = wmma_bf16(aq1, b11, s1);
        }
        for (int r = 0; r < 8; r++) {
            float v = fmaxf(s0[r], s1[r]);
            for (int off = 1; off < 16; off <<= 1) v = fmaxf(v, __shfl_xor(v, off, 32));
            float mnew = fmaxf(mrow[r], v);
            float alpha = __expf(mrow[r] - mnew);
            mrow[r] = mnew;
            float p0 = __expf(s0[r] - mnew);
            float p1 = __expf(s1[r] - mnew);
            s0[r] = p0; s1[r] = p1;
            float ps = p0 + p1;
            for (int off = 1; off < 16; off <<= 1) ps += __shfl_xor(ps, off, 32);
            lrow[r] = lrow[r] * alpha + ps;
            acc[0][r] *= alpha; acc[1][r] *= alpha; acc[2][r] *= alpha; acc[3][r] *= alpha;
        }
        __syncthreads();
        for (int r = 0; r < 8; r++) {
            int row = r + hi * 8;
            lds_p[row * 32 + r15]      = f2bf(s0[r]);
            lds_p[row * 32 + 16 + r15] = f2bf(s1[r]);
        }
        __syncthreads();
        v16bf pa = frag_a(lds_p, 32, lane);
        for (int j = 0; j < 4; j++) {
            v16bf vb = frag_b(vbase + (long)(j * 16) * Sn + kc, Sn, lane);
            acc[j] = wmma_bf16(pa, vb, acc[j]);
        }
    }
    for (int j = 0; j < 4; j++) {
        int col = nh * Dn + j * 16 + r15;
        for (int r = 0; r < 8; r++) {
            int row = t0 + r + hi * 8;
            ob[(long)row * Hn + col] = f2bf(acc[j][r] / lrow[r]);
        }
    }
}

// -------- LN2 fused with gate logits + argmax + per-expert token bucketing --------
__global__ void k_ln2_gate(const float* __restrict__ x1, const float* __restrict__ g,
                           const float* __restrict__ bt, const float* __restrict__ gw,
                           const float* __restrict__ gb, unsigned short* __restrict__ hn2,
                           int* __restrict__ counts, int* __restrict__ toks) {
    int t = blockIdx.x, tid = threadIdx.x;
    const float* row = x1 + (long)t * Hn;
    float4 v = ((const float4*)row)[tid];
    __shared__ float r1[256], r2[256];
    __shared__ float logit[En];
    r1[tid] = v.x + v.y + v.z + v.w;
    r2[tid] = v.x * v.x + v.y * v.y + v.z * v.z + v.w * v.w;
    __syncthreads();
    for (int st = 128; st > 0; st >>= 1) {
        if (tid < st) { r1[tid] += r1[tid + st]; r2[tid] += r2[tid + st]; }
        __syncthreads();
    }
    float mu = r1[0] / Hn;
    float inv = rsqrtf(r2[0] / Hn - mu * mu + EPSF);
    __syncthreads();
    int h0 = tid * 4;
    float n[4];
    n[0] = (v.x - mu) * inv * g[h0 + 0] + bt[h0 + 0];
    n[1] = (v.y - mu) * inv * g[h0 + 1] + bt[h0 + 1];
    n[2] = (v.z - mu) * inv * g[h0 + 2] + bt[h0 + 2];
    n[3] = (v.w - mu) * inv * g[h0 + 3] + bt[h0 + 3];
    ushort4 p; p.x = f2bf(n[0]); p.y = f2bf(n[1]); p.z = f2bf(n[2]); p.w = f2bf(n[3]);
    *(ushort4*)&hn2[(long)t * Hn + h0] = p;
    float lg[En];
    for (int e = 0; e < En; e++) lg[e] = 0.f;
    for (int j = 0; j < 4; j++) {
        const float* grow = gw + (long)(h0 + j) * En;
        for (int e = 0; e < En; e++) lg[e] += n[j] * grow[e];
    }
    for (int e = 0; e < En; e++) {
        r1[tid] = lg[e];
        __syncthreads();
        for (int st = 128; st > 0; st >>= 1) {
            if (tid < st) r1[tid] += r1[tid + st];
            __syncthreads();
        }
        if (tid == 0) logit[e] = r1[0] + gb[e];
        __syncthreads();
    }
    if (tid == 0) {
        int best = 0; float bvv = logit[0];
        for (int e = 1; e < En; e++) if (logit[e] > bvv) { bvv = logit[e]; best = e; }
        int slot = atomicAdd(&counts[best], 1);
        toks[best * Tn + slot] = t;
    }
}

// -------- fused top-1 MoE FFN: one (expert, 16-token tile) per wave, mid kept in 128KB LDS --------
// Activation staging uses the CDNA5 async global->LDS path (ASYNCcnt).
__global__ __launch_bounds__(32) void k_moe(const unsigned short* __restrict__ hn2,
                                            const unsigned short* __restrict__ w1t,
                                            const unsigned short* __restrict__ w2t,
                                            const float* __restrict__ b1,
                                            const float* __restrict__ b2,
                                            const float* __restrict__ x1,
                                            const int* __restrict__ counts,
                                            const int* __restrict__ toks,
                                            float* __restrict__ out) {
    __shared__ __align__(16) unsigned short lds_a[16 * Hn];     // 32 KB gathered activations
    __shared__ __align__(16) unsigned short lds_mid[16 * Fn];   // 128 KB intermediate
    __shared__ int stok[16];
    int lane = threadIdx.x;
    int e = blockIdx.y, tile = blockIdx.x;
    int nt = counts[e];
    int t0 = tile * 16;
    if (t0 >= nt) return;
    if (lane < 16) {
        int idx = t0 + lane; if (idx >= nt) idx = nt - 1;
        stok[lane] = toks[e * Tn + idx];
    }
    __syncthreads();
    // ---- async global->LDS staging of 16 gathered activation rows (32 KB) ----
    // LDS destination offset = low 32 bits of the generic pointer (LDS aperture truncation).
    for (int row = 0; row < 16; ++row) {
        const unsigned short* src = hn2 + (long)stok[row] * Hn;
        for (int c = lane * 8; c < Hn; c += 32 * 8) {
            unsigned int ldsoff = (unsigned int)(unsigned long long)&lds_a[row * Hn + c];
            const void* gp = (const void*)(src + c);
            asm volatile("global_load_async_to_lds_b128 %0, %1, off"
                         :
                         : "v"(ldsoff), "v"(gp)
                         : "memory");
        }
    }
    asm volatile("s_wait_asynccnt 0" ::: "memory");
    __syncthreads();
    const unsigned short* w1e = w1t + (long)e * Fn * Hn;  // [F,H] bf16
    const unsigned short* w2e = w2t + (long)e * Hn * Fn;  // [H,F] bf16
    int r15 = lane & 15, hi = (lane >> 4) & 1;
    for (int nf = 0; nf < Fn; nf += 16) {
        v8f c = {};
        const unsigned short* bb = w1e + (long)nf * Hn;
        for (int k = 0; k < Hn; k += 32) {
            v16bf a = frag_a(lds_a + k, Hn, lane);
            v16bf b = frag_b(bb + k, Hn, lane);
            c = wmma_bf16(a, b, c);
        }
        int col = nf + r15;
        float bv = b1[e * Fn + col];
        for (int r = 0; r < 8; r++) {
            float val = c[r] + bv;
            val = val > 0.f ? val : 0.f;  // relu
            lds_mid[(r + hi * 8) * Fn + col] = f2bf(val);
        }
    }
    __syncthreads();
    for (int nh2 = 0; nh2 < Hn; nh2 += 16) {
        v8f c = {};
        const unsigned short* bb = w2e + (long)nh2 * Fn;
        for (int k = 0; k < Fn; k += 32) {
            v16bf a = frag_a(lds_mid + k, Fn, lane);
            v16bf b = frag_b(bb + k, Fn, lane);
            c = wmma_bf16(a, b, c);
        }
        int col = nh2 + r15;
        float bv = b2[e * Hn + col];
        for (int r = 0; r < 8; r++) {
            int m = r + hi * 8;
            if (t0 + m < nt) {
                long t = stok[m];
                out[t * Hn + col] = x1[t * Hn + col] + c[r] + bv;
            }
        }
    }
}

extern "C" void kernel_launch(void* const* d_in, const int* in_sizes, int n_in,
                              void* d_out, int out_size, void* d_ws, size_t ws_size,
                              hipStream_t stream) {
    (void)in_sizes; (void)n_in; (void)out_size; (void)ws_size;
    const float* x    = (const float*)d_in[0];
    const float* ln1g = (const float*)d_in[1];
    const float* ln1b = (const float*)d_in[2];
    const float* ln2g = (const float*)d_in[3];
    const float* ln2b = (const float*)d_in[4];
    const float* wq = (const float*)d_in[5];  const float* bq = (const float*)d_in[6];
    const float* wk = (const float*)d_in[7];  const float* bk = (const float*)d_in[8];
    const float* wv = (const float*)d_in[9];  const float* bv = (const float*)d_in[10];
    const float* wo = (const float*)d_in[11]; const float* bo = (const float*)d_in[12];
    const float* gw = (const float*)d_in[13]; const float* gb = (const float*)d_in[14];
    const float* w1 = (const float*)d_in[15]; const float* b1 = (const float*)d_in[16];
    const float* w2 = (const float*)d_in[17]; const float* b2 = (const float*)d_in[18];
    float* out = (float*)d_out;

    char* ws = (char*)d_ws;
    size_t off = 0;
    auto alloc = [&](size_t bytes) -> char* {
        char* p = ws + off;
        off = (off + bytes + 255) & ~(size_t)255;
        return p;
    };
    unsigned short* hn1  = (unsigned short*)alloc((size_t)Tn * Hn * 2);
    unsigned short* wqT  = (unsigned short*)alloc((size_t)Hn * Hn * 2);
    unsigned short* wkT  = (unsigned short*)alloc((size_t)Hn * Hn * 2);
    unsigned short* wvT  = (unsigned short*)alloc((size_t)Hn * Hn * 2);
    unsigned short* woT  = (unsigned short*)alloc((size_t)Hn * Hn * 2);
    unsigned short* qb   = (unsigned short*)alloc((size_t)Tn * Hn * 2);
    unsigned short* kbuf = (unsigned short*)alloc((size_t)Tn * Hn * 2);
    unsigned short* vtb  = (unsigned short*)alloc((size_t)Tn * Hn * 2);
    unsigned short* obuf = (unsigned short*)alloc((size_t)Tn * Hn * 2);
    float*          x1   = (float*)alloc((size_t)Tn * Hn * 4);
    unsigned short* hn2  = (unsigned short*)alloc((size_t)Tn * Hn * 2);
    unsigned short* w1t  = (unsigned short*)alloc((size_t)En * Fn * Hn * 2);
    unsigned short* w2t  = (unsigned short*)alloc((size_t)En * Hn * Fn * 2);
    int*            cnts = (int*)alloc(En * sizeof(int));
    int*            toks = (int*)alloc((size_t)En * Tn * sizeof(int));

    hipMemsetAsync(cnts, 0, En * sizeof(int), stream);

    dim3 tb(32, 8);
    k_transpose_cvt<<<dim3(Hn / 32, Hn / 32, 1), tb, 0, stream>>>(wq, wqT, Hn, Hn);
    k_transpose_cvt<<<dim3(Hn / 32, Hn / 32, 1), tb, 0, stream>>>(wk, wkT, Hn, Hn);
    k_transpose_cvt<<<dim3(Hn / 32, Hn / 32, 1), tb, 0, stream>>>(wv, wvT, Hn, Hn);
    k_transpose_cvt<<<dim3(Hn / 32, Hn / 32, 1), tb, 0, stream>>>(wo, woT, Hn, Hn);
    k_transpose_cvt<<<dim3(Hn / 32, Fn / 32, En), tb, 0, stream>>>(w1, w1t, Hn, Fn);
    k_transpose_cvt<<<dim3(Fn / 32, Hn / 32, En), tb, 0, stream>>>(w2, w2t, Fn, Hn);

    k_ln_bf16<<<Tn, 256, 0, stream>>>(x, ln1g, ln1b, hn1);

    dim3 gg(Tn / 32, Hn / 32);
    k_gemm<0><<<gg, 32, 0, stream>>>(hn1, wqT, bq, nullptr, 0.125f, qb, nullptr, Tn, Hn, Hn);
    k_gemm<0><<<gg, 32, 0, stream>>>(hn1, wkT, bk, nullptr, 1.0f, kbuf, nullptr, Tn, Hn, Hn);
    k_gemm<1><<<gg, 32, 0, stream>>>(hn1, wvT, bv, nullptr, 1.0f, vtb, nullptr, Tn, Hn, Hn);

    k_attn<<<dim3(Sn / 16, NHn, Bn), 32, 0, stream>>>(qb, kbuf, vtb, obuf);

    k_gemm<2><<<gg, 32, 0, stream>>>(obuf, woT, bo, x, 1.0f, nullptr, x1, Tn, Hn, Hn);

    k_ln2_gate<<<Tn, 256, 0, stream>>>(x1, ln2g, ln2b, gw, gb, hn2, cnts, toks);

    k_moe<<<dim3(Tn / 16, En), 32, 0, stream>>>(hn2, w1t, w2t, b1, b2, x1, cnts, toks, out);
}
